// MultiHeadAttentionLayer_5342939316533
// MI455X (gfx1250) — matrix operand order repaired and checked by reading, not compiled, not run on hardware
//
#include <hip/hip_runtime.h>
#include <hip/hip_bf16.h>
#include <cmath>

#define NUM_HEAD 16
#define EMB 1024
#define DK 64
#define SEQ 2048
#define BATCH 2
#define LN_EPS 1e-5f

typedef __attribute__((ext_vector_type(16))) _Float16 v16h;
typedef __attribute__((ext_vector_type(8)))  float    v8f;
typedef __attribute__((ext_vector_type(4)))  unsigned int v4u;
typedef __attribute__((ext_vector_type(8)))  int      v8i;
typedef __attribute__((ext_vector_type(4)))  int      v4i;

// ---------------------------------------------------------------------------
// Tensor Data Mover: 2-D tile load Global -> LDS (descriptor per ISA 8.3-8.6).
//   elem_code: 0=1B, 1=2B, 2=4B, 3=8B
//   dim0/dim1: tensor extents (elements);  stride0: row stride (elements)
//   tile0/tile1: tile extents (elements / rows)
// Issued by one wave; completion via s_wait_tensorcnt.
// (6-arg builtin form: g0, g1, g2, g3, g4, cpol)
// ---------------------------------------------------------------------------
__device__ __forceinline__ void tdm_load_2d(unsigned int lds_off,
                                            const void* gptr,
                                            unsigned int elem_code,
                                            unsigned int dim0, unsigned int dim1,
                                            unsigned int stride0,
                                            unsigned int tile0, unsigned int tile1) {
  const unsigned long long ga = (unsigned long long)gptr;
  v4u g0;
  g0[0] = 1u;                                          // count=1, user mode
  g0[1] = lds_off;                                     // lds_addr (bytes)
  g0[2] = (unsigned int)(ga & 0xffffffffull);          // global_addr[31:0]
  g0[3] = (unsigned int)((ga >> 32) & 0x1ffffffull)    // global_addr[56:32]
        | 0x80000000u;                                 // type=2 ("image")
  v8i g1;
  g1[0] = (int)(elem_code << 16);                      // data_size; wg_mask=0
  g1[1] = (int)((dim0 & 0xffffu) << 16);               // tensor_dim0 lo
  g1[2] = (int)(((dim0 >> 16) & 0xffffu) | ((dim1 & 0xffffu) << 16));
  g1[3] = (int)(((dim1 >> 16) & 0xffffu) | ((tile0 & 0xffffu) << 16));
  g1[4] = (int)(tile1 & 0xffffu);                      // tile_dim1; tile_dim2=0
  g1[5] = (int)stride0;                                // tensor_dim0_stride lo
  g1[6] = 0;                                           // stride0 hi | stride1 lo
  g1[7] = 0;
  const v4i z4 = {0, 0, 0, 0};
  const v8i z8 = {0, 0, 0, 0, 0, 0, 0, 0};
  __builtin_amdgcn_tensor_load_to_lds(g0, g1, z4, z4, z8, 0);
}

__device__ __forceinline__ unsigned int lds_offset(const void* p) {
  return (unsigned int)(unsigned long long)p;          // low 32 bits = LDS offset
}

// ---------------------------------------------------------------------------
// WMMA fragment loaders (wave32; A/B use contiguous k = 16*(lane/16) + s, not the ISA table order)
// A (16x32 f16, MxK): lanes 0-15 -> M=lane, VGPR j holds K=2j,2j+1 ;
//                     lanes 16-31 -> same M, K offset +16.
// B (32x16 f16, KxN): lane%16 = N ; VGPR j holds K=2j,2j+1 (lanes<16),
//                     K=16+2j,16+2j+1 (lanes>=16).
// C/D (16x16 f32):    lane%16 = N ; VGPR i -> M=i (lanes<16), M=8+i (lanes>=16).
// ---------------------------------------------------------------------------
__device__ __forceinline__ v16h load_a_f32(const float* src, int ld) {
  const int lane = threadIdx.x & 31;
  const int row  = lane & 15;
  const int hi   = lane >> 4;
  v16h a;
#pragma unroll
  for (int j = 0; j < 8; ++j) {
    const int kb = 2 * j + 16 * hi;
    const float* p = src + row * ld + kb;
    a[2 * j + 0] = (_Float16)p[0];
    a[2 * j + 1] = (_Float16)p[1];
  }
  return a;
}

__device__ __forceinline__ v16h load_a_f16(const _Float16* src, int ld) {
  const int lane = threadIdx.x & 31;
  const int row  = lane & 15;
  const int hi   = lane >> 4;
  v16h a;
#pragma unroll
  for (int j = 0; j < 8; ++j) {
    const int kb = 2 * j + 16 * hi;
    const _Float16* p = src + row * ld + kb;
    a[2 * j + 0] = p[0];
    a[2 * j + 1] = p[1];
  }
  return a;
}

__device__ __forceinline__ v16h load_b_f32(const float* src, int ld) {
  const int lane = threadIdx.x & 31;
  const int col  = lane & 15;
  const int hi   = lane >> 4;
  v16h b;
#pragma unroll
  for (int j = 0; j < 8; ++j) {
    const int k = 2 * j + 16 * hi;
    b[2 * j + 0] = (_Float16)src[(k + 0) * ld + col];
    b[2 * j + 1] = (_Float16)src[(k + 1) * ld + col];
  }
  return b;
}

__device__ __forceinline__ v16h load_b_f16(const _Float16* src, int ld) {
  const int lane = threadIdx.x & 31;
  const int col  = lane & 15;
  const int hi   = lane >> 4;
  v16h b;
#pragma unroll
  for (int j = 0; j < 8; ++j) {
    const int k = 2 * j + 16 * hi;
    b[2 * j + 0] = src[(k + 0) * ld + col];
    b[2 * j + 1] = src[(k + 1) * ld + col];
  }
  return b;
}

// B fragment from a row-major matrix that is logically transposed:
// element (k, n) = src[n * ld + k]   (used for K^T in the score GEMM)
__device__ __forceinline__ v16h load_bT_f16(const _Float16* src, int ld) {
  const int lane = threadIdx.x & 31;
  const int col  = lane & 15;
  const int hi   = lane >> 4;
  v16h b;
#pragma unroll
  for (int j = 0; j < 8; ++j) {
    const int k = 2 * j + 16 * hi;
    b[2 * j + 0] = src[col * ld + k + 0];
    b[2 * j + 1] = src[col * ld + k + 1];
  }
  return b;
}

#define WMMA_F16(a, b, c) \
  __builtin_amdgcn_wmma_f32_16x16x32_f16(false, (a), false, (b), (short)0, (c), false, false)

// ---------------------------------------------------------------------------
// Kernel 1: QKV projections.  O = X @ W + bias, output f16 in [B,H,S,64].
// Block = 8 waves, 16x128 output tile.  A stripe (16x1024 fp32) staged once
// into LDS by the Tensor Data Mover, reused by all 8 waves (ds reads).
// grid = (M/16=256, N/128=8, 3), block = 256.
// ---------------------------------------------------------------------------
__global__ __launch_bounds__(256) void qkv_proj_kernel(
    const float* __restrict__ xq, const float* __restrict__ xk,
    const float* __restrict__ xv, const float* __restrict__ wq,
    const float* __restrict__ wk, const float* __restrict__ wv,
    const float* __restrict__ bq, const float* __restrict__ bk,
    const float* __restrict__ bv, _Float16* __restrict__ q_ws,
    _Float16* __restrict__ k_ws, _Float16* __restrict__ v_ws) {
  __shared__ float sA[16 * EMB];  // 64 KB A stripe

  const int which = blockIdx.z;
  const float* x = (which == 0) ? xq : (which == 1) ? xk : xv;
  const float* w = (which == 0) ? wq : (which == 1) ? wk : wv;
  const float* bias = (which == 0) ? bq : (which == 1) ? bk : bv;
  _Float16* dst = (which == 0) ? q_ws : (which == 1) ? k_ws : v_ws;

  const int m0 = blockIdx.x * 16;
  const int wave = threadIdx.x >> 5;
  const int lane = threadIdx.x & 31;
  const int n0 = blockIdx.y * 128 + wave * 16;

  if (wave == 0) {
    tdm_load_2d(lds_offset(sA), x + (size_t)m0 * EMB, /*f32*/ 2,
                EMB, BATCH * SEQ, EMB, EMB, 16);
    __builtin_amdgcn_s_wait_tensorcnt(0);
  }
  __syncthreads();

  v8f c = {};
  for (int kk = 0; kk < EMB; kk += 32) {
    if (kk + 64 < EMB)
      __builtin_prefetch(w + (size_t)(kk + 64) * EMB + n0, 0, 1);
    v16h a = load_a_f32(&sA[kk], EMB);                 // ds reads
    v16h b = load_b_f32(w + (size_t)kk * EMB + n0, EMB);
    c = WMMA_F16(a, b, c);
  }

  const int n = n0 + (lane & 15);
  const int head = n >> 6;  // 64 columns per head
  const int d = n & 63;
  const float bval = bias[n];
  const int mh = (lane >> 4) * 8;
#pragma unroll
  for (int i = 0; i < 8; ++i) {
    const int m = m0 + mh + i;
    const int b_ = m >> 11;  // / 2048
    const int s = m & 2047;
    dst[(((size_t)(b_ * NUM_HEAD + head)) * SEQ + s) * DK + d] =
        (_Float16)(c[i] + bval);
  }
}

// ---------------------------------------------------------------------------
// Kernel 2: attention.  One block = 16 query rows of one (b,h).
// Q tile TDM-staged into LDS.  Phase 1: scores (16x2048) in LDS via WMMA.
// Phase 2: softmax in LDS, fp32 probs to d_out (head-major).  Phase 3:
// ctx = P @ V via WMMA.  grid = (128, 32), block = 256.  LDS = 138 KB.
// ---------------------------------------------------------------------------
__global__ __launch_bounds__(256) void attn_kernel(
    const _Float16* __restrict__ q_ws, const _Float16* __restrict__ k_ws,
    const _Float16* __restrict__ v_ws, const unsigned char* __restrict__ mask,
    float* __restrict__ attn_out, _Float16* __restrict__ ctx_ws) {
  __shared__ float sc[16 * SEQ];       // 128 KB score stripe
  __shared__ float ctxp[2 * 16 * 64];  // 8 KB partial context
  __shared__ _Float16 sq[16 * DK];     // 2 KB Q tile

  const int qt = blockIdx.x;
  const int bh = blockIdx.y;  // b * NUM_HEAD + h
  const int b = bh >> 4;
  const int h = bh & 15;
  const _Float16* qh = q_ws + (size_t)bh * SEQ * DK;
  const _Float16* kh = k_ws + (size_t)bh * SEQ * DK;
  const _Float16* vh = v_ws + (size_t)bh * SEQ * DK;

  const int wave = threadIdx.x >> 5;
  const int lane = threadIdx.x & 31;
  const int q0 = qt * 16;

  if (wave == 0) {
    tdm_load_2d(lds_offset(sq), qh + (size_t)q0 * DK, /*f16*/ 1,
                DK, SEQ, DK, DK, 16);
    __builtin_amdgcn_s_wait_tensorcnt(0);
  }
  __syncthreads();

  // Q fragments (dk = 64 -> two 32-wide k-steps), from LDS, hoisted.
  const v16h a0 = load_a_f16(&sq[0], DK);
  const v16h a1 = load_a_f16(&sq[32], DK);

  // ---- Phase 1: scores -> LDS ----
  for (int ct = wave; ct < SEQ / 16; ct += 8) {
    const int c0 = ct * 16;
    v8f acc = {};
    v16h b0 = load_bT_f16(kh + (size_t)c0 * DK + 0, DK);
    acc = WMMA_F16(a0, b0, acc);
    v16h b1 = load_bT_f16(kh + (size_t)c0 * DK + 32, DK);
    acc = WMMA_F16(a1, b1, acc);

    const int n = c0 + (lane & 15);
    const int mh = (lane >> 4) * 8;
#pragma unroll
    for (int i = 0; i < 8; ++i) {
      const int m = mh + i;
      float sv = acc[i] * 0.125f;  // 1/sqrt(dk)
      if (mask[((size_t)b * SEQ + (q0 + m)) * SEQ + n]) sv = -INFINITY;
      sc[m * SEQ + n] = sv;
    }
  }
  __syncthreads();

  // ---- Phase 2: softmax (one wave per row) + write probs ----
  for (int r = wave; r < 16; r += 8) {
    float mx = -INFINITY;
    for (int cc = lane; cc < SEQ; cc += 32) mx = fmaxf(mx, sc[r * SEQ + cc]);
#pragma unroll
    for (int off = 16; off > 0; off >>= 1) mx = fmaxf(mx, __shfl_xor(mx, off, 32));

    float sum = 0.f;
    for (int cc = lane; cc < SEQ; cc += 32) {
      const float e = __expf(sc[r * SEQ + cc] - mx);
      sc[r * SEQ + cc] = e;
      sum += e;
    }
#pragma unroll
    for (int off = 16; off > 0; off >>= 1) sum += __shfl_xor(sum, off, 32);
    const float inv = 1.0f / sum;

    float* aout = attn_out + (((size_t)h * BATCH + b) * SEQ + (q0 + r)) * SEQ;
    for (int cc = lane; cc < SEQ; cc += 32) {
      const float p = sc[r * SEQ + cc] * inv;
      sc[r * SEQ + cc] = p;
      aout[cc] = p;
    }
  }
  __syncthreads();

  // ---- Phase 3: ctx = P @ V  (4 dv-tiles x 2 K-halves across 8 waves) ----
  {
    const int ntile = wave & 3;
    const int khalf = wave >> 2;
    const int d0 = ntile * 16;
    v8f acc = {};
    for (int kk = khalf * 1024; kk < khalf * 1024 + 1024; kk += 32) {
      v16h a = load_a_f32(&sc[kk], SEQ);  // probs from LDS -> f16
      v16h bb = load_b_f16(vh + (size_t)kk * DK + d0, DK);
      acc = WMMA_F16(a, bb, acc);
    }
    const int n = d0 + (lane & 15);
    const int mh = (lane >> 4) * 8;
#pragma unroll
    for (int i = 0; i < 8; ++i)
      ctxp[khalf * 1024 + (mh + i) * 64 + n] = acc[i];
  }
  __syncthreads();

  // Combine the two K-half partials, store ctx as f16 in [B*S, H*64].
  for (int idx = threadIdx.x; idx < 16 * 64; idx += 256) {
    const int m = idx >> 6;
    const int d = idx & 63;
    const float v = ctxp[idx] + ctxp[1024 + idx];
    ctx_ws[((size_t)(b * SEQ + q0 + m)) * EMB + h * DK + d] = (_Float16)v;
  }
}

// ---------------------------------------------------------------------------
// Kernel 3: out = LayerNorm(ctx @ w_fc + b_fc + input_q).
// ctx stripe (16x1024 f16) TDM-staged into LDS; 8 waves x 8 N-tiles; fused LN.
// grid = 256, block = 256.  LDS = 96 KB.
// ---------------------------------------------------------------------------
__global__ __launch_bounds__(256) void fc_ln_kernel(
    const _Float16* __restrict__ ctx_ws, const float* __restrict__ w_fc,
    const float* __restrict__ b_fc, const float* __restrict__ xq,
    const float* __restrict__ gamma, const float* __restrict__ beta,
    float* __restrict__ out) {
  __shared__ float y[16 * EMB];        // 64 KB
  __shared__ _Float16 sctx[16 * EMB];  // 32 KB A stripe

  const int m0 = blockIdx.x * 16;
  const int wave = threadIdx.x >> 5;
  const int lane = threadIdx.x & 31;

  if (wave == 0) {
    tdm_load_2d(lds_offset(sctx), ctx_ws + (size_t)m0 * EMB, /*f16*/ 1,
                EMB, BATCH * SEQ, EMB, EMB, 16);
    __builtin_amdgcn_s_wait_tensorcnt(0);
  }
  __syncthreads();

  v8f acc[8] = {};
  for (int kk = 0; kk < EMB; kk += 32) {
    if (kk + 64 < EMB)
      __builtin_prefetch(w_fc + (size_t)(kk + 64) * EMB + wave * 128, 0, 1);
    v16h a = load_a_f16(&sctx[kk], EMB);  // ds reads
#pragma unroll
    for (int t = 0; t < 8; ++t) {
      const int n0 = (wave * 8 + t) * 16;
      v16h bb = load_b_f32(w_fc + (size_t)kk * EMB + n0, EMB);
      acc[t] = WMMA_F16(a, bb, acc[t]);
    }
  }

  // bias + residual into LDS
#pragma unroll
  for (int t = 0; t < 8; ++t) {
    const int n0 = (wave * 8 + t) * 16;
    const int n = n0 + (lane & 15);
    const int mh = (lane >> 4) * 8;
    const float bfc = b_fc[n];
#pragma unroll
    for (int i = 0; i < 8; ++i) {
      const int m = mh + i;
      y[m * EMB + n] = acc[t][i] + bfc + xq[(size_t)(m0 + m) * EMB + n];
    }
  }
  __syncthreads();

  // fused LayerNorm: one wave per row
  for (int r = wave; r < 16; r += 8) {
    float s = 0.f, s2 = 0.f;
    for (int cc = lane; cc < EMB; cc += 32) {
      const float v = y[r * EMB + cc];
      s += v;
      s2 += v * v;
    }
#pragma unroll
    for (int off = 16; off > 0; off >>= 1) {
      s += __shfl_xor(s, off, 32);
      s2 += __shfl_xor(s2, off, 32);
    }
    const float mu = s * (1.0f / EMB);
    const float var = s2 * (1.0f / EMB) - mu * mu;  // biased variance
    const float rinv = rsqrtf(var + LN_EPS);
    const size_t row = (size_t)(m0 + r);
    for (int cc = lane; cc < EMB; cc += 32)
      out[row * EMB + cc] = (y[r * EMB + cc] - mu) * rinv * gamma[cc] + beta[cc];
  }
}

// ---------------------------------------------------------------------------
// Launch
// ---------------------------------------------------------------------------
extern "C" void kernel_launch(void* const* d_in, const int* in_sizes, int n_in,
                              void* d_out, int out_size, void* d_ws,
                              size_t ws_size, hipStream_t stream) {
  const float* xq = (const float*)d_in[0];
  const float* xk = (const float*)d_in[1];
  const float* xv = (const float*)d_in[2];
  const unsigned char* mask = (const unsigned char*)d_in[3];
  const float* wq = (const float*)d_in[4];
  const float* bq = (const float*)d_in[5];
  const float* wk = (const float*)d_in[6];
  const float* bk = (const float*)d_in[7];
  const float* wv = (const float*)d_in[8];
  const float* bv = (const float*)d_in[9];
  const float* wfc = (const float*)d_in[10];
  const float* bfc = (const float*)d_in[11];
  const float* gamma = (const float*)d_in[12];
  const float* beta = (const float*)d_in[13];

  float* outp = (float*)d_out;                      // [B,Sq,EMB]
  float* attnp = outp + (size_t)BATCH * SEQ * EMB;  // [H*B,Sq,Skv]

  const size_t QKV_BYTES = (size_t)BATCH * SEQ * EMB * sizeof(_Float16);  // 8 MB
  char* ws = (char*)d_ws;
  _Float16* q_ws = (_Float16*)(ws + 0 * QKV_BYTES);
  _Float16* k_ws = (_Float16*)(ws + 1 * QKV_BYTES);
  _Float16* v_ws = (_Float16*)(ws + 2 * QKV_BYTES);
  _Float16* ctx_ws = (_Float16*)(ws + 3 * QKV_BYTES);

  {
    dim3 grid(BATCH * SEQ / 16, EMB / 128, 3);
    qkv_proj_kernel<<<grid, 256, 0, stream>>>(xq, xk, xv, wq, wk, wv, bq, bk,
                                              bv, q_ws, k_ws, v_ws);
  }
  {
    dim3 grid(SEQ / 16, BATCH * NUM_HEAD);
    attn_kernel<<<grid, 256, 0, stream>>>(q_ws, k_ws, v_ws, mask, attnp,
                                          ctx_ws);
  }
  {
    fc_ln_kernel<<<BATCH * SEQ / 16, 256, 0, stream>>>(ctx_ws, wfc, bfc, xq,
                                                       gamma, beta, outp);
  }
}
